// SEModel_6459630813828
// MI455X (gfx1250) — matrix operand order repaired and checked by reading, not compiled, not run on hardware
//
#include <hip/hip_runtime.h>
#include <hip/hip_bf16.h>

#define NN_TOT   16384
#define DIM      256
#define EDIM     128
#define KCL      16
#define KNN      32
#define NPAIRS   (NN_TOT * KNN)
#define ALPHA    0.3f
#define N_ITERS  5
#define EPSF     1e-12f
#define WAVES    8

typedef __attribute__((ext_vector_type(16))) _Float16 v16h;
typedef __attribute__((ext_vector_type(8)))  _Float16 v8h;
typedef __attribute__((ext_vector_type(8)))  float    v8f;

#define CAT16(lo, hi) __builtin_shufflevector(lo, hi, 0,1,2,3,4,5,6,7,8,9,10,11,12,13,14,15)

__device__ __forceinline__ v8f wmma_f16(v16h a, v16h b, v8f c) {
  // 8 args: (neg_a, A, neg_b, B, c_mod, C, reuse_a, reuse_b)
  return __builtin_amdgcn_wmma_f32_16x16x32_f16(false, a, false, b, (short)0, c, false, false);
}

// ---------------------------------------------------------------- utilities
__global__ void k_zero(float* __restrict__ p, int n) {
  int i = blockIdx.x * blockDim.x + threadIdx.x;
  if (i < n) p[i] = 0.f;
}

__global__ void k_cast_f16(const float* __restrict__ s, _Float16* __restrict__ d, int n) {
  int i = blockIdx.x * blockDim.x + threadIdx.x;
  if (i < n) d[i] = (_Float16)s[i];
}

__global__ void k_relu(float* __restrict__ p, int n) {
  int i = blockIdx.x * blockDim.x + threadIdx.x;
  if (i < n) p[i] = fmaxf(p[i], 0.f);
}

// Row-normalize x (for cosine KNN) and stage both normalized and raw copies as f16.
__global__ __launch_bounds__(256) void k_normalize(const float* __restrict__ x,
                                                   _Float16* __restrict__ xh,
                                                   _Float16* __restrict__ xrh) {
  __shared__ float red[256];
  const int tid = threadIdx.x;
  const size_t base = (size_t)blockIdx.x * DIM;
  float v = x[base + tid];
  red[tid] = v * v;
  __syncthreads();
  for (int s = 128; s > 0; s >>= 1) {
    if (tid < s) red[tid] += red[tid + s];
    __syncthreads();
  }
  float nrm = sqrtf(red[0]) + EPSF;
  xh[base + tid]  = (_Float16)(v / nrm);
  xrh[base + tid] = (_Float16)v;
}

// ------------------------------------------------ fused WMMA KNN + top-32
// Block = 16 rows, 8 waves. Per column stripe of 128 nodes:
//   phase 1: each wave does 8x v_wmma_f32_16x16x32_f16 (B panel held in VGPRs),
//            then immediately issues next stripe's B loads (overlap with scan),
//   phase 2: 256 threads filter 2048 candidates against per-row threshold,
//            pushing survivors to an LDS queue (ds atomics),
//   phase 3: 16 row-owner lanes merge the few survivors into their top-32.
__global__ __launch_bounds__(256) void k_knn(const _Float16* __restrict__ xh,
                                             int* __restrict__ knn_idx,
                                             float* __restrict__ knn_w) {
  __shared__ float ctile[WAVES][256];
  __shared__ float topv[16][KNN];
  __shared__ int   topi[16][KNN];
  __shared__ float rowmin[16];
  __shared__ int   ccnt[16];
  __shared__ float candv[16][128];
  __shared__ int   candc[16][128];

  const int tid  = threadIdx.x;
  const int wave = tid >> 5;
  const int lane = tid & 31;
  const int m    = lane & 15;
  const int hi   = lane >> 4;      // half-lane select
  const int rb   = blockIdx.x * 16;

  // Preload all A fragments for this row block (K = 256 -> 8 frags kept in VGPRs)
  v16h afrag[8];
  const _Float16* arow = xh + (size_t)(rb + m) * DIM;
#pragma unroll
  for (int kk = 0; kk < 8; ++kk) {
    const int kb = kk * 32;
    v8h lo = *(const v8h*)(arow + kb + hi * 16);
    v8h hh = *(const v8h*)(arow + kb + hi * 16 + 8);
    afrag[kk] = CAT16(lo, hh);
  }
  if (tid < 16) {
#pragma unroll
    for (int j = 0; j < KNN; ++j) { topv[tid][j] = -1e30f; topi[tid][j] = -1; }
    rowmin[tid] = -1e30f;
    ccnt[tid] = 0;
  }

  // Preload the first B panel into registers.
  v16h bfr[8];
  {
    const _Float16* brow = xh + (size_t)(wave * 16 + m) * DIM + hi * 16;
#pragma unroll
    for (int kk = 0; kk < 8; ++kk) bfr[kk] = *(const v16h*)(brow + kk * 32);
  }
  __syncthreads();

  float minv = -1e30f;   // valid for tid < 16 (row owners)
  int   minpos = 0;

  for (int cbase = 0; cbase < NN_TOT; cbase += 16 * WAVES) {
    // ---- phase 1: matrix math on the resident B panel
    v8f c = {};
#pragma unroll
    for (int kk = 0; kk < 8; ++kk) c = wmma_f16(afrag[kk], bfr[kk], c);

    // Software pipeline: issue next stripe's B loads now; they complete while
    // the candidate-filter/merge phases run below.
    const int nbase = cbase + 16 * WAVES;
    if (nbase < NN_TOT) {
      const _Float16* brow = xh + (size_t)(nbase + wave * 16 + m) * DIM + hi * 16;
#pragma unroll
      for (int kk = 0; kk < 8; ++kk) bfr[kk] = *(const v16h*)(brow + kk * 32);
    }

#pragma unroll
    for (int v = 0; v < 8; ++v) ctile[wave][(v + hi * 8) * 16 + m] = c[v];
    __syncthreads();

    // ---- phase 2: parallel threshold filter (16 threads per row)
    {
      const int r    = tid >> 4;
      const int j    = tid & 15;
      const int grow = rb + r;
      const float thr = rowmin[r];
      for (int cc2 = j; cc2 < 16 * WAVES; cc2 += 16) {
        const int w  = cc2 >> 4;
        const int nn = cc2 & 15;
        const float val = ctile[w][r * 16 + nn];
        const int col = cbase + cc2;            // == cbase + w*16 + nn
        if (col != grow && val > thr) {
          int pos = atomicAdd(&ccnt[r], 1);
          candv[r][pos] = val;
          candc[r][pos] = col;
        }
      }
    }
    __syncthreads();

    // ---- phase 3: row owners merge survivors into top-32
    if (tid < 16) {
      const int en = ccnt[tid];
      for (int e = 0; e < en; ++e) {
        const float val = candv[tid][e];
        if (val > minv) {
          topv[tid][minpos] = val;
          topi[tid][minpos] = candc[tid][e];
          minv = topv[tid][0]; minpos = 0;
#pragma unroll
          for (int j = 1; j < KNN; ++j) {
            float tv = topv[tid][j];
            if (tv < minv) { minv = tv; minpos = j; }
          }
        }
      }
      ccnt[tid]   = 0;
      rowmin[tid] = minv;
    }
    __syncthreads();
  }

  if (tid < 16) {
    const int grow = rb + tid;
#pragma unroll
    for (int j = 0; j < KNN; ++j) {
      float v = fminf(fmaxf(topv[tid][j], 0.f), 1.f);   // clip(cos, 0, 1)
      knn_idx[(size_t)grow * KNN + j] = topi[tid][j];
      knn_w[(size_t)grow * KNN + j]  = v;
    }
  }
}

// ----------------------------------------------- generic WMMA GEMM: C = A*B^T + bias
// A: [rows x K] f16 row-major (lda), B: [Ncols x K] f16 row-major (ldb), C f32 (ldc).
template <int K>
__global__ __launch_bounds__(256) void k_gemm_bias(const _Float16* __restrict__ A, int lda,
                                                   const _Float16* __restrict__ B, int ldb,
                                                   const float* __restrict__ bias,
                                                   float* __restrict__ C, int ldc,
                                                   int Ncols) {
  constexpr int KF = K / 32;
  const int tid  = threadIdx.x;
  const int wave = tid >> 5;
  const int lane = tid & 31;
  const int m    = lane & 15;
  const int hi   = lane >> 4;
  const int rb   = blockIdx.x * 16;

  v16h afrag[KF];
  const _Float16* arow = A + (size_t)(rb + m) * lda;
#pragma unroll
  for (int kk = 0; kk < KF; ++kk) {
    const int kb = kk * 32;
    v8h lo = *(const v8h*)(arow + kb + hi * 16);
    v8h hh = *(const v8h*)(arow + kb + hi * 16 + 8);
    afrag[kk] = CAT16(lo, hh);
  }

  for (int cb = wave * 16; cb < Ncols; cb += 16 * WAVES) {
    v8f c = {};
    const _Float16* brow = B + (size_t)(cb + m) * ldb + hi * 16;
#pragma unroll
    for (int kk = 0; kk < KF; ++kk) {
      v16h b = *(const v16h*)(brow + kk * 32);
      c = wmma_f16(afrag[kk], b, c);
    }
    const int col = cb + m;
#pragma unroll
    for (int v = 0; v < 8; ++v) {
      const int M = v + hi * 8;
      C[(size_t)(rb + M) * ldc + col] = c[v] + bias[col];
    }
  }
}

// ---------------------------------------------- edge (pair) scatter kernels
__global__ void k_deg(const int* __restrict__ idx, const float* __restrict__ w,
                      float* __restrict__ deg) {
  int p = blockIdx.x * blockDim.x + threadIdx.x;
  if (p >= NPAIRS) return;
  int i = p >> 5;
  int n = idx[p];
  float v = 0.5f * w[p];
  atomicAdd(&deg[i], v);
  atomicAdd(&deg[n], v);
}

template <int F>
__global__ __launch_bounds__(256) void k_conv(const int* __restrict__ idx,
                                              const float* __restrict__ w,
                                              const float* __restrict__ deg,
                                              const float* __restrict__ hin,
                                              float* __restrict__ hout) {
  const int tid = threadIdx.x;
  const int ppb = 256 / F;
  int p = blockIdx.x * ppb + tid / F;
  int f = tid % F;
  if (p >= NPAIRS) return;
  int i = p >> 5;
  int n = idx[p];
  float v  = 0.5f * w[p];
  float di = fmaxf(deg[i], EPSF);
  float dn = fmaxf(deg[n], EPSF);
  atomicAdd(&hout[(size_t)n * F + f], hin[(size_t)i * F + f] * (v / dn));
  atomicAdd(&hout[(size_t)i * F + f], hin[(size_t)n * F + f] * (v / di));
}

__global__ __launch_bounds__(256) void k_as(const int* __restrict__ idx,
                                            const float* __restrict__ w,
                                            const float* __restrict__ S,
                                            float* __restrict__ AS) {
  const int tid = threadIdx.x;
  int p = blockIdx.x * 16 + tid / 16;
  int f = tid % 16;
  if (p >= NPAIRS) return;
  int i = p >> 5;
  int n = idx[p];
  float v = 0.5f * w[p];
  atomicAdd(&AS[(size_t)i * 16 + f], v * S[(size_t)n * 16 + f]);
  atomicAdd(&AS[(size_t)n * 16 + f], v * S[(size_t)i * 16 + f]);
}

__global__ void k_msg(const int* __restrict__ idx, const float* __restrict__ w,
                      const float* __restrict__ t, float* __restrict__ msg) {
  int p = blockIdx.x * blockDim.x + threadIdx.x;
  if (p >= NPAIRS) return;
  int i = p >> 5;
  int n = idx[p];
  float v = 0.5f * w[p];
  atomicAdd(&msg[i], v * t[n]);
  atomicAdd(&msg[n], v * t[i]);
}

// ------------------------------------------------------ row softmax (16 cols)
__global__ void k_softmax16(float* __restrict__ S) {
  int r = blockIdx.x * blockDim.x + threadIdx.x;
  if (r >= NN_TOT) return;
  float s[16];
  float mx = -1e30f;
#pragma unroll
  for (int f = 0; f < 16; ++f) { s[f] = S[(size_t)r * 16 + f]; mx = fmaxf(mx, s[f]); }
  float sum = 0.f;
#pragma unroll
  for (int f = 0; f < 16; ++f) { s[f] = expf(s[f] - mx); sum += s[f]; }
  float inv = 1.f / sum;
#pragma unroll
  for (int f = 0; f < 16; ++f) S[(size_t)r * 16 + f] = s[f] * inv;
}

// --------------------------------------------- A_c = S^T @ AS (16x16 outputs)
__global__ __launch_bounds__(256) void k_Ac(const float* __restrict__ S,
                                            const float* __restrict__ AS,
                                            float* __restrict__ Ac) {
  __shared__ float red[256];
  const int tid = threadIdx.x;
  const int a = blockIdx.x >> 4, b = blockIdx.x & 15;
  float acc = 0.f;
  for (int n = tid; n < NN_TOT; n += 256)
    acc += S[(size_t)n * 16 + a] * AS[(size_t)n * 16 + b];
  red[tid] = acc;
  __syncthreads();
  for (int s = 128; s > 0; s >>= 1) {
    if (tid < s) red[tid] += red[tid + s];
    __syncthreads();
  }
  if (tid == 0) Ac[a * 16 + b] = red[0];
}

// ----------------- tiny 16x16 spectral step: Laplacian + Jacobi eigh -> tau
__global__ void k_spectral(const float* __restrict__ Ac, float* __restrict__ tau) {
  if (threadIdx.x != 0 || blockIdx.x != 0) return;
  float A[KCL][KCL], L[KCL][KCL], V[KCL][KCL];
  for (int a = 0; a < KCL; ++a)
    for (int b = 0; b < KCL; ++b) {
      float v = 0.5f * (Ac[a * KCL + b] + Ac[b * KCL + a]);
      A[a][b] = (a == b) ? 0.f : fmaxf(v, 0.f);
    }
  float isq[KCL];
  for (int a = 0; a < KCL; ++a) {
    float s = 0.f;
    for (int b = 0; b < KCL; ++b) s += A[a][b];
    isq[a] = 1.f / sqrtf(s + EPSF);
  }
  for (int a = 0; a < KCL; ++a)
    for (int b = 0; b < KCL; ++b) {
      L[a][b] = ((a == b) ? 1.f : 0.f) - isq[a] * A[a][b] * isq[b];
      V[a][b] = (a == b) ? 1.f : 0.f;
    }
  // cyclic Jacobi
  for (int sweep = 0; sweep < 30; ++sweep) {
    for (int p = 0; p < KCL - 1; ++p)
      for (int q = p + 1; q < KCL; ++q) {
        float apq = L[p][q];
        if (fabsf(apq) < 1e-13f) continue;
        float theta = (L[q][q] - L[p][p]) / (2.f * apq);
        float tt = 1.f / (fabsf(theta) + sqrtf(theta * theta + 1.f));
        if (theta < 0.f) tt = -tt;
        float cc = 1.f / sqrtf(tt * tt + 1.f);
        float ss = tt * cc;
        for (int r = 0; r < KCL; ++r) {
          float lrp = L[r][p], lrq = L[r][q];
          L[r][p] = cc * lrp - ss * lrq;
          L[r][q] = ss * lrp + cc * lrq;
        }
        for (int c2 = 0; c2 < KCL; ++c2) {
          float lpc = L[p][c2], lqc = L[q][c2];
          L[p][c2] = cc * lpc - ss * lqc;
          L[q][c2] = ss * lpc + cc * lqc;
        }
        for (int r = 0; r < KCL; ++r) {
          float vrp = V[r][p], vrq = V[r][q];
          V[r][p] = cc * vrp - ss * vrq;
          V[r][q] = ss * vrp + cc * vrq;
        }
      }
  }
  float ev[KCL];
  for (int r = 0; r < KCL; ++r) ev[r] = L[r][r];
  int i0 = 0;
  for (int r = 1; r < KCL; ++r) if (ev[r] < ev[i0]) i0 = r;
  int i1 = (i0 == 0) ? 1 : 0;
  for (int r = 0; r < KCL; ++r) if (r != i0 && ev[r] < ev[i1]) i1 = r;
  float mean = 0.f;
  for (int r = 0; r < KCL; ++r) mean += V[r][i1];
  mean *= (1.f / KCL);
  float var = 0.f;
  for (int r = 0; r < KCL; ++r) { float d = V[r][i1] - mean; var += d * d; }
  var *= (1.f / (KCL - 1));          // ddof = 1
  float sd = sqrtf(var) + EPSF;
  for (int r = 0; r < KCL; ++r) tau[r] = (V[r][i1] - mean) / sd;
}

// ---------------------------------------------- propagation-loop kernels
__global__ void k_t0(const float* __restrict__ S, const float* __restrict__ tau,
                     float* __restrict__ t) {
  int n = blockIdx.x * blockDim.x + threadIdx.x;
  if (n >= NN_TOT) return;
  float acc = 0.f;
#pragma unroll
  for (int f = 0; f < 16; ++f) acc += S[(size_t)n * 16 + f] * tau[f];
  t[n] = acc;
}

__global__ __launch_bounds__(256) void k_coldot(const float* __restrict__ S,
                                                const float* __restrict__ t,
                                                float* __restrict__ out16) {
  __shared__ float part[16];
  const int tid = threadIdx.x;
  if (tid < 16) part[tid] = 0.f;
  __syncthreads();
  const int f  = tid & 15;
  const int n0 = blockIdx.x * 16 + (tid >> 4);
  float acc = 0.f;
  for (int n = n0; n < NN_TOT; n += gridDim.x * 16)
    acc += S[(size_t)n * 16 + f] * (t ? t[n] : 1.f);
  atomicAdd(&part[f], acc);
  __syncthreads();
  if (tid < 16) atomicAdd(&out16[tid], part[tid]);
}

__global__ void k_tnew(const float* __restrict__ t, const float* __restrict__ msg,
                       const float* __restrict__ deg, float* __restrict__ tnew) {
  int n = blockIdx.x * blockDim.x + threadIdx.x;
  if (n >= NN_TOT) return;
  float nbr = msg[n] / (deg[n] + EPSF);
  tnew[n] = (1.f - ALPHA) * t[n] + ALPHA * nbr;
}

__global__ void k_delta(const float* __restrict__ mbnum, const float* __restrict__ mcnum,
                        const float* __restrict__ sumS, float* __restrict__ delta) {
  int f = threadIdx.x;
  if (f < 16) delta[f] = (mbnum[f] - mcnum[f]) / (sumS[f] + EPSF);
}

__global__ void k_update(const float* __restrict__ S, const float* __restrict__ tnew,
                         const float* __restrict__ delta, float* __restrict__ tout) {
  int n = blockIdx.x * blockDim.x + threadIdx.x;
  if (n >= NN_TOT) return;
  float acc = tnew[n];
#pragma unroll
  for (int f = 0; f < 16; ++f) acc += S[(size_t)n * 16 + f] * delta[f];
  tout[n] = acc;
}

// =================================================================== launch
extern "C" void kernel_launch(void* const* d_in, const int* in_sizes, int n_in,
                              void* d_out, int out_size, void* d_ws, size_t ws_size,
                              hipStream_t stream) {
  (void)in_sizes; (void)n_in; (void)out_size; (void)ws_size;
  const float* x   = (const float*)d_in[0];
  const float* W_g = (const float*)d_in[1];
  const float* b_g = (const float*)d_in[2];
  // d_in[3] (W_e), d_in[4] (b_e): `h` is dead in the reference forward -> skipped.
  const float* W_a = (const float*)d_in[5];
  const float* b_a = (const float*)d_in[6];

  char* wp = (char*)d_ws;
  auto alloc = [&](size_t bytes) -> void* {
    void* r = (void*)wp;
    wp += (bytes + 255) & ~(size_t)255;
    return r;
  };
  _Float16* xh   = (_Float16*)alloc((size_t)NN_TOT * DIM * 2);
  _Float16* xrh  = (_Float16*)alloc((size_t)NN_TOT * DIM * 2);
  _Float16* wgh  = (_Float16*)alloc((size_t)EDIM * DIM * 2);
  _Float16* wah  = (_Float16*)alloc((size_t)KCL * EDIM * 2);
  _Float16* h0h  = (_Float16*)alloc((size_t)NN_TOT * EDIM * 2);
  float* g     = (float*)alloc((size_t)NN_TOT * EDIM * 4);
  float* h0    = (float*)alloc((size_t)NN_TOT * EDIM * 4);
  float* alin  = (float*)alloc((size_t)NN_TOT * KCL * 4);
  float* slog  = (float*)alloc((size_t)NN_TOT * KCL * 4);   // becomes S after softmax
  float* AS    = (float*)alloc((size_t)NN_TOT * KCL * 4);
  int*   kidx  = (int*)alloc((size_t)NPAIRS * 4);
  float* kw    = (float*)alloc((size_t)NPAIRS * 4);
  float* deg   = (float*)alloc((size_t)NN_TOT * 4);
  float* t     = (float*)alloc((size_t)NN_TOT * 4);
  float* tnew  = (float*)alloc((size_t)NN_TOT * 4);
  float* msg   = (float*)alloc((size_t)NN_TOT * 4);
  float* Ac    = (float*)alloc(256 * 4);
  float* tau   = (float*)alloc(16 * 4);
  float* sumS  = (float*)alloc(16 * 4);
  float* mbnum = (float*)alloc(16 * 4);
  float* mcnum = (float*)alloc(16 * 4);
  float* delta = (float*)alloc(16 * 4);

  const dim3 B(256);
  // stage inputs to f16
  k_normalize<<<NN_TOT, B, 0, stream>>>(x, xh, xrh);
  k_cast_f16<<<(EDIM * DIM + 255) / 256, B, 0, stream>>>(W_g, wgh, EDIM * DIM);
  k_cast_f16<<<(KCL * EDIM + 255) / 256, B, 0, stream>>>(W_a, wah, KCL * EDIM);

  // KNN graph (fused WMMA GEMM + top-32)
  k_knn<<<NN_TOT / 16, B, 0, stream>>>(xh, kidx, kw);
  k_zero<<<(NN_TOT + 255) / 256, B, 0, stream>>>(deg, NN_TOT);
  k_deg<<<(NPAIRS + 255) / 256, B, 0, stream>>>(kidx, kw, deg);

  // h0 = relu(conv(x @ W_g^T + b_g))
  k_gemm_bias<DIM><<<NN_TOT / 16, B, 0, stream>>>(xrh, DIM, wgh, DIM, b_g, g, EDIM, EDIM);
  k_zero<<<(NN_TOT * EDIM + 255) / 256, B, 0, stream>>>(h0, NN_TOT * EDIM);
  k_conv<EDIM><<<NPAIRS / (256 / EDIM), B, 0, stream>>>(kidx, kw, deg, g, h0);
  k_relu<<<(NN_TOT * EDIM + 255) / 256, B, 0, stream>>>(h0, NN_TOT * EDIM);
  k_cast_f16<<<(NN_TOT * EDIM + 255) / 256, B, 0, stream>>>(h0, h0h, NN_TOT * EDIM);

  // S = softmax(relu(conv(h0 @ W_a^T + b_a)))
  k_gemm_bias<EDIM><<<NN_TOT / 16, B, 0, stream>>>(h0h, EDIM, wah, EDIM, b_a, alin, KCL, KCL);
  k_zero<<<(NN_TOT * KCL + 255) / 256, B, 0, stream>>>(slog, NN_TOT * KCL);
  k_conv<KCL><<<NPAIRS / (256 / KCL), B, 0, stream>>>(kidx, kw, deg, alin, slog);
  k_relu<<<(NN_TOT * KCL + 255) / 256, B, 0, stream>>>(slog, NN_TOT * KCL);
  k_softmax16<<<(NN_TOT + 255) / 256, B, 0, stream>>>(slog);

  // A_c = S^T @ (w-weighted S scatter); spectral tau
  k_zero<<<(NN_TOT * KCL + 255) / 256, B, 0, stream>>>(AS, NN_TOT * KCL);
  k_as<<<NPAIRS / 16, B, 0, stream>>>(kidx, kw, slog, AS);
  k_Ac<<<256, B, 0, stream>>>(slog, AS, Ac);
  k_spectral<<<1, 32, 0, stream>>>(Ac, tau);

  // t0 = S @ tau ; sumS ; mean_back numerator
  k_t0<<<(NN_TOT + 255) / 256, B, 0, stream>>>(slog, tau, t);
  k_zero<<<1, B, 0, stream>>>(sumS, 16);
  k_coldot<<<64, B, 0, stream>>>(slog, nullptr, sumS);
  k_zero<<<1, B, 0, stream>>>(mbnum, 16);
  k_coldot<<<64, B, 0, stream>>>(slog, t, mbnum);

  // 5 propagation iterations
  for (int it = 0; it < N_ITERS; ++it) {
    k_zero<<<(NN_TOT + 255) / 256, B, 0, stream>>>(msg, NN_TOT);
    k_msg<<<(NPAIRS + 255) / 256, B, 0, stream>>>(kidx, kw, t, msg);
    k_tnew<<<(NN_TOT + 255) / 256, B, 0, stream>>>(t, msg, deg, tnew);
    k_zero<<<1, B, 0, stream>>>(mcnum, 16);
    k_coldot<<<64, B, 0, stream>>>(slog, tnew, mcnum);
    k_delta<<<1, 16, 0, stream>>>(mbnum, mcnum, sumS, delta);
    float* tout = (it == N_ITERS - 1) ? (float*)d_out : t;
    k_update<<<(NN_TOT + 255) / 256, B, 0, stream>>>(slog, tnew, delta, tout);
  }
}
